// ConformerEncoderLayer_58660663329432
// MI455X (gfx1250) — compile-verified, matrix operand order checked by reading, not run on hardware
//
#include <hip/hip_runtime.h>

// ---------------------------------------------------------------------------
// Conformer encoder layer for MI455X (gfx1250), bf16 WMMA pipeline with
// double-buffered async global->LDS tile staging (ASYNCcnt path).
// ---------------------------------------------------------------------------

typedef __bf16 bf16_t;
typedef __bf16 v16bf __attribute__((ext_vector_type(16)));
typedef float  v8f   __attribute__((ext_vector_type(8)));

constexpr int T_   = 1024;
constexpr int B_   = 4;
constexpr int E_   = 512;
constexpr int H_   = 8;
constexpr int D_   = 64;
constexpr int DFF_ = 2048;
constexpr int KC_  = 31;
constexpr int TB_  = T_ * B_;          // 4096 token rows
constexpr int E3_  = 3 * E_;           // 1536
constexpr int PR_  = 2 * T_ - 1;       // 2047 rel-pos rows

__device__ __forceinline__ float dswish(float x) {
    // x * sigmoid(x - 1) == x / (1 + e^(1-x))
    return x / (1.0f + __expf(1.0f - x));
}
__device__ __forceinline__ float sigmoidf(float x) {
    return 1.0f / (1.0f + __expf(-x));
}

// ---------------------------------------------------------------------------
// Generic bf16 WMMA GEMM:  C[M,N] = act( A[M,K] * B[N,K]^T + bias[N] ) + resid
// Block: 256 threads = 8 wave32, block tile 64(M) x 128(N), K step 32.
// Each wave computes a 32x32 tile as 2x2 WMMA 16x16x32_bf16 accumulators.
// Tiles are staged with GLOBAL_LOAD_ASYNC_TO_LDS_B128 (ASYNCcnt), double
// buffered so the next tile's DMA overlaps the current tile's WMMAs.
// OOB tile rows are address-clamped (their results are masked at store time),
// so every lane issues its async copy unconditionally -- no EXEC branching.
// Requires: lda, ldb multiples of 8; A/B base 16B aligned; K % 32 == 0.
// ---------------------------------------------------------------------------
union FragBF { v16bf v; uint4 q[2]; };

__global__ void __launch_bounds__(256)
wmma_gemm_kernel(const bf16_t* __restrict__ A, int lda,
                 const bf16_t* __restrict__ Bm, int ldb,
                 const float* __restrict__ bias,
                 const float* __restrict__ resid, int ldr,
                 float* __restrict__ Cf, int ldcf,
                 bf16_t* __restrict__ Ch, int ldch,
                 int M, int N, int KD, int act)
{
    __shared__ __attribute__((aligned(16))) bf16_t As[2][64][40];   // 32 cols + pad
    __shared__ __attribute__((aligned(16))) bf16_t Bs[2][128][40];

    const int tid  = threadIdx.x;
    const int wave = tid >> 5;
    const int lane = tid & 31;
    const int l    = lane & 15;     // column-within-tile lane
    const int hf   = lane >> 4;     // K-octet selector / M-half selector
    const int wm   = wave >> 2;     // 0..1
    const int wn   = wave & 3;      // 0..3
    const int mBase = blockIdx.y * 64;
    const int nBase = blockIdx.x * 128;

    v8f acc[2][2];
#pragma unroll
    for (int i = 0; i < 2; ++i)
#pragma unroll
        for (int j = 0; j < 2; ++j)
#pragma unroll
            for (int e = 0; e < 8; ++e) acc[i][j][e] = 0.0f;

    // Issue this thread's async global->LDS copies for K-tile k0 into buffer b.
    auto stageTile = [&](int buf, int k0) {
        {   // A tile: 64 rows x 32 cols = 256 x 16B chunks (1 per thread)
            const int row = tid >> 2;
            const int kc  = (tid & 3) * 8;
            int gr = mBase + row;
            if (gr >= M) gr = M - 1;               // clamp: OOB rows never stored
            const bf16_t* gp = A + (size_t)gr * lda + k0 + kc;
            const unsigned lp = (unsigned)(size_t)&As[buf][row][kc];
            asm volatile("global_load_async_to_lds_b128 %0, %1, off"
                         :: "v"(lp), "v"(gp) : "memory");
        }
#pragma unroll
        for (int c = 0; c < 2; ++c) {              // B tile: 512 chunks (2/thread)
            const int cid = tid + c * 256;
            const int row = cid >> 2;
            const int kc  = (cid & 3) * 8;
            int gn = nBase + row;
            if (gn >= N) gn = N - 1;               // clamp: OOB cols never stored
            const bf16_t* gp = Bm + (size_t)gn * ldb + k0 + kc;
            const unsigned lp = (unsigned)(size_t)&Bs[buf][row][kc];
            asm volatile("global_load_async_to_lds_b128 %0, %1, off"
                         :: "v"(lp), "v"(gp) : "memory");
        }
    };

    // Prologue: fill buffer 0.
    stageTile(0, 0);
    asm volatile("s_wait_asynccnt 0x0" ::: "memory");
    __syncthreads();

    int buf = 0;
    for (int k0 = 0; k0 < KD; k0 += 32, buf ^= 1) {
        // Kick off DMA for the next K-tile into the other buffer.
        if (k0 + 32 < KD)
            stageTile(buf ^ 1, k0 + 32);

        // Fragment loads per documented CDNA5 16-bit A/B lane layout:
        // lane holds K = hf*8 + [0..7] and K = 16 + hf*8 + [0..7].
        FragBF fa[2], fb[2];
#pragma unroll
        for (int i = 0; i < 2; ++i) {
            const int r = wm * 32 + i * 16 + l;
            fa[i].q[0] = *(const uint4*)(&As[buf][r][hf * 8]);
            fa[i].q[1] = *(const uint4*)(&As[buf][r][16 + hf * 8]);
        }
#pragma unroll
        for (int j = 0; j < 2; ++j) {
            const int r = wn * 32 + j * 16 + l;
            fb[j].q[0] = *(const uint4*)(&Bs[buf][r][hf * 8]);
            fb[j].q[1] = *(const uint4*)(&Bs[buf][r][16 + hf * 8]);
        }
#pragma unroll
        for (int i = 0; i < 2; ++i)
#pragma unroll
            for (int j = 0; j < 2; ++j)
                acc[i][j] = __builtin_amdgcn_wmma_f32_16x16x32_bf16(
                    false, fa[i].v, false, fb[j].v,
                    (short)0, acc[i][j], false, false);

        // This wave's async copies done; then workgroup-wide visibility.
        asm volatile("s_wait_asynccnt 0x0" ::: "memory");
        __syncthreads();
    }

    // Epilogue: C/D layout -> local M = v + 8*hf, local N = l.
#pragma unroll
    for (int i = 0; i < 2; ++i)
#pragma unroll
        for (int j = 0; j < 2; ++j) {
            const int gc = nBase + wn * 32 + j * 16 + l;
            if (gc >= N) continue;
            const float bb = bias ? bias[gc] : 0.0f;
#pragma unroll
            for (int v = 0; v < 8; ++v) {
                const int gr = mBase + wm * 32 + i * 16 + hf * 8 + v;
                if (gr >= M) continue;
                float val = acc[i][j][v] + bb;
                if (act == 1) val = dswish(val);
                if (resid) val += resid[(size_t)gr * ldr + gc];
                if (Cf) Cf[(size_t)gr * ldcf + gc] = val;
                if (Ch) Ch[(size_t)gr * ldch + gc] = (bf16_t)val;
            }
        }
}

// ---------------------------------------------------------------------------
// Elementwise / helper kernels
// ---------------------------------------------------------------------------
__global__ void cvt_f32_bf16_kernel(const float* __restrict__ in,
                                    bf16_t* __restrict__ out, int n)
{
    int i = blockIdx.x * 256 + threadIdx.x;
    if (i < n) out[i] = (bf16_t)in[i];
}

// qkv (T,B,3E) f32 -> per-head Qu/Qv/K (B,H,T,D) bf16 and Vt (B,H,D,T) bf16
__global__ void repack_qkv_kernel(const float* __restrict__ qkv,
                                  const float* __restrict__ u,
                                  const float* __restrict__ v,
                                  bf16_t* __restrict__ Qu,
                                  bf16_t* __restrict__ Qv,
                                  bf16_t* __restrict__ Kh,
                                  bf16_t* __restrict__ Vt)
{
    int i = blockIdx.x * 256 + threadIdx.x;
    if (i >= B_ * H_ * T_ * D_) return;
    const int d = i % D_;
    const int t = (i / D_) % T_;
    const int h = (i / (D_ * T_)) % H_;
    const int b = i / (D_ * T_ * H_);
    const size_t base = ((size_t)t * B_ + b) * E3_ + h * D_ + d;
    const float q  = qkv[base] * 0.125f;          // D^{-1/2}, D = 64
    const float kk = qkv[base + E_];
    const float vv = qkv[base + 2 * E_];
    const size_t bh = (size_t)b * H_ + h;
    const size_t o  = (bh * T_ + t) * D_ + d;
    Qu[o] = (bf16_t)(q + u[h * D_ + d]);
    Qv[o] = (bf16_t)(q + v[h * D_ + d]);
    Kh[o] = (bf16_t)kk;
    Vt[(bh * D_ + d) * T_ + t] = (bf16_t)vv;      // transposed for attn*V GEMM
}

// scores[t,j] = ac[t,j] + bd[t, (T-1)-t+j]; softmax over j; bf16 probs out.
__global__ void relshift_softmax_kernel(const float* __restrict__ ac,
                                        const float* __restrict__ bd,
                                        bf16_t* __restrict__ probs)
{
    const int t   = blockIdx.x;
    const int tid = threadIdx.x;
    __shared__ float red[256];

    float vals[4];
    float m = -1e30f;
#pragma unroll
    for (int c = 0; c < 4; ++c) {
        const int j = tid + c * 256;
        const float v = ac[(size_t)t * T_ + j]
                      + bd[(size_t)t * PR_ + (T_ - 1 - t + j)];
        vals[c] = v;
        m = fmaxf(m, v);
    }
    red[tid] = m;
    __syncthreads();
    for (int s = 128; s > 0; s >>= 1) {
        if (tid < s) red[tid] = fmaxf(red[tid], red[tid + s]);
        __syncthreads();
    }
    const float mx = red[0];
    __syncthreads();

    float sum = 0.0f;
#pragma unroll
    for (int c = 0; c < 4; ++c) {
        vals[c] = __expf(vals[c] - mx);
        sum += vals[c];
    }
    red[tid] = sum;
    __syncthreads();
    for (int s = 128; s > 0; s >>= 1) {
        if (tid < s) red[tid] += red[tid + s];
        __syncthreads();
    }
    const float inv = 1.0f / red[0];
#pragma unroll
    for (int c = 0; c < 4; ++c) {
        const int j = tid + c * 256;
        probs[(size_t)t * T_ + j] = (bf16_t)(vals[c] * inv);
    }
}

// G (TB, 2E) f32 -> GLU -> glu (B*E, T) f32   (channel-major for depthwise conv)
__global__ void glu_kernel(const float* __restrict__ G, float* __restrict__ glu)
{
    int i = blockIdx.x * 256 + threadIdx.x;
    if (i >= TB_ * E_) return;
    const int c = i % E_;
    const int r = i / E_;       // r = t*B + b
    const int b = r % B_;
    const int t = r / B_;
    const float a = G[(size_t)r * (2 * E_) + c];
    const float g = G[(size_t)r * (2 * E_) + E_ + c];
    glu[((size_t)b * E_ + c) * T_ + t] = a * sigmoidf(g);
}

// Depthwise conv (K=31, pad 15) + bias + DoubleSwish; writes bf16 (TB,E).
__global__ void dwconv_kernel(const float* __restrict__ glu,
                              const float* __restrict__ w,
                              const float* __restrict__ bconv,
                              bf16_t* __restrict__ outY)
{
    __shared__ float row[T_ + KC_ - 1];    // 1054
    const int bc = blockIdx.x;             // b*E + c
    const int c  = bc % E_;
    const int b  = bc / E_;
    const float* in = glu + (size_t)bc * T_;

    for (int idx = threadIdx.x; idx < T_ + KC_ - 1; idx += 256) {
        const int tp = idx - (KC_ - 1) / 2;
        row[idx] = (tp >= 0 && tp < T_) ? in[tp] : 0.0f;
    }
    __syncthreads();

    float wv[KC_];
#pragma unroll
    for (int k = 0; k < KC_; ++k) wv[k] = w[c * KC_ + k];
    const float bb = bconv[c];

    for (int t = threadIdx.x; t < T_; t += 256) {
        float s = 0.0f;
#pragma unroll
        for (int k = 0; k < KC_; ++k) s += row[t + k] * wv[k];
        const float val = dswish(s + bb);
        outY[((size_t)t * B_ + b) * E_ + c] = (bf16_t)val;
    }
}

// BasicNorm: y = x * rsqrt(mean(x^2) + exp(log_eps)); one block per row.
__global__ void basicnorm_kernel(const float* __restrict__ x,
                                 const float* __restrict__ log_eps,
                                 float* __restrict__ y)
{
    const int r   = blockIdx.x;
    const int tid = threadIdx.x;
    __shared__ float red[256];
    float s = 0.0f;
    for (int c = tid; c < E_; c += 256) {
        const float v = x[(size_t)r * E_ + c];
        s += v * v;
    }
    red[tid] = s;
    __syncthreads();
    for (int st = 128; st > 0; st >>= 1) {
        if (tid < st) red[tid] += red[tid + st];
        __syncthreads();
    }
    const float scale = rsqrtf(red[0] / (float)E_ + __expf(log_eps[0]));
    for (int c = tid; c < E_; c += 256)
        y[(size_t)r * E_ + c] = x[(size_t)r * E_ + c] * scale;
}

// ---------------------------------------------------------------------------
// Host launcher
// ---------------------------------------------------------------------------
extern "C" void kernel_launch(void* const* d_in, const int* in_sizes, int n_in,
                              void* d_out, int out_size, void* d_ws, size_t ws_size,
                              hipStream_t stream)
{
    (void)in_sizes; (void)n_in; (void)out_size; (void)ws_size;

    // Inputs in setup_inputs() order (all f32)
    const float* src        = (const float*)d_in[0];
    const float* pos_emb    = (const float*)d_in[1];
    const float* in_proj_w  = (const float*)d_in[2];
    const float* in_proj_b  = (const float*)d_in[3];
    const float* out_proj_w = (const float*)d_in[4];
    const float* out_proj_b = (const float*)d_in[5];
    const float* lin_pos_w  = (const float*)d_in[6];
    const float* pos_u      = (const float*)d_in[7];
    const float* pos_v      = (const float*)d_in[8];
    const float* ffm_w1     = (const float*)d_in[9];
    const float* ffm_b1     = (const float*)d_in[10];
    const float* ffm_w2     = (const float*)d_in[11];
    const float* ffm_b2     = (const float*)d_in[12];
    const float* ff_w1      = (const float*)d_in[13];
    const float* ff_b1      = (const float*)d_in[14];
    const float* ff_w2      = (const float*)d_in[15];
    const float* ff_b2      = (const float*)d_in[16];
    const float* pw1_w      = (const float*)d_in[17];
    const float* pw1_b      = (const float*)d_in[18];
    const float* dw_w       = (const float*)d_in[19];
    const float* dw_b       = (const float*)d_in[20];
    const float* pw2_w      = (const float*)d_in[21];
    const float* pw2_b      = (const float*)d_in[22];
    const float* norm_eps   = (const float*)d_in[23];
    float* out = (float*)d_out;

    // Bump allocator over workspace (256B aligned)
    char* base = (char*)d_ws;
    size_t cur = 0;
    auto alloc = [&](size_t bytes) -> void* {
        cur = (cur + 255) & ~(size_t)255;
        void* p = base + cur;
        cur += bytes;
        return p;
    };
    auto allocH = [&](size_t n) { return (bf16_t*)alloc(n * sizeof(bf16_t)); };
    auto allocF = [&](size_t n) { return (float*)alloc(n * sizeof(float)); };

    // bf16 weight / activation copies
    bf16_t* srch   = allocH((size_t)TB_ * E_);
    bf16_t* posh   = allocH((size_t)PR_ * E_);
    bf16_t* ffm1h  = allocH((size_t)DFF_ * E_);
    bf16_t* ffm2h  = allocH((size_t)E_ * DFF_);
    bf16_t* inh    = allocH((size_t)E3_ * E_);
    bf16_t* outh   = allocH((size_t)E_ * E_);
    bf16_t* wposh  = allocH((size_t)E_ * E_);
    bf16_t* ff1h   = allocH((size_t)DFF_ * E_);
    bf16_t* ff2h   = allocH((size_t)E_ * DFF_);
    bf16_t* pw1h   = allocH((size_t)2 * E_ * E_);
    bf16_t* pw2h   = allocH((size_t)E_ * E_);

    float*  X1     = allocF((size_t)TB_ * E_);
    bf16_t* X1h    = allocH((size_t)TB_ * E_);
    bf16_t* Hh     = allocH((size_t)TB_ * DFF_);     // reused by both FFNs
    float*  QKV    = allocF((size_t)TB_ * E3_);
    bf16_t* Quh    = allocH((size_t)B_ * H_ * T_ * D_);
    bf16_t* Qvh    = allocH((size_t)B_ * H_ * T_ * D_);
    bf16_t* Khh    = allocH((size_t)B_ * H_ * T_ * D_);
    bf16_t* Vth    = allocH((size_t)B_ * H_ * D_ * T_);
    bf16_t* Ph     = allocH((size_t)PR_ * E_);
    float*  acB    = allocF((size_t)T_ * T_);        // reused per (b,h)
    float*  bdB    = allocF((size_t)T_ * PR_);       // reused per (b,h)
    bf16_t* prB    = allocH((size_t)T_ * T_);        // reused per (b,h)
    bf16_t* Ctxh   = allocH((size_t)TB_ * E_);
    float*  X2     = allocF((size_t)TB_ * E_);
    bf16_t* X2h    = allocH((size_t)TB_ * E_);
    float*  Gbuf   = allocF((size_t)TB_ * 2 * E_);
    float*  gluB   = allocF((size_t)B_ * E_ * T_);
    bf16_t* Y2h    = allocH((size_t)TB_ * E_);
    float*  X3     = allocF((size_t)TB_ * E_);
    bf16_t* X3h    = allocH((size_t)TB_ * E_);
    float*  X4     = allocF((size_t)TB_ * E_);

    auto cvt = [&](const float* in, bf16_t* o, size_t n) {
        cvt_f32_bf16_kernel<<<dim3((unsigned)((n + 255) / 256)), 256, 0, stream>>>(in, o, (int)n);
    };
    auto gemm = [&](const bf16_t* A, int lda, const bf16_t* Bm, int ldb,
                    const float* bias, const float* res, int ldr,
                    float* Cf, int ldcf, bf16_t* Ch, int ldch,
                    int M, int N, int KD, int act) {
        dim3 grid((N + 127) / 128, (M + 63) / 64);
        wmma_gemm_kernel<<<grid, 256, 0, stream>>>(A, lda, Bm, ldb, bias, res, ldr,
                                                   Cf, ldcf, Ch, ldch, M, N, KD, act);
    };

    // --- bf16 conversions -------------------------------------------------
    cvt(src,        srch,  (size_t)TB_ * E_);
    cvt(pos_emb,    posh,  (size_t)PR_ * E_);
    cvt(ffm_w1,     ffm1h, (size_t)DFF_ * E_);
    cvt(ffm_w2,     ffm2h, (size_t)E_ * DFF_);
    cvt(in_proj_w,  inh,   (size_t)E3_ * E_);
    cvt(out_proj_w, outh,  (size_t)E_ * E_);
    cvt(lin_pos_w,  wposh, (size_t)E_ * E_);
    cvt(ff_w1,      ff1h,  (size_t)DFF_ * E_);
    cvt(ff_w2,      ff2h,  (size_t)E_ * DFF_);
    cvt(pw1_w,      pw1h,  (size_t)2 * E_ * E_);
    cvt(pw2_w,      pw2h,  (size_t)E_ * E_);

    // --- macaron FFN ------------------------------------------------------
    gemm(srch, E_, ffm1h, E_, ffm_b1, nullptr, 0,
         nullptr, 0, Hh, DFF_, TB_, DFF_, E_, /*dswish*/1);
    gemm(Hh, DFF_, ffm2h, DFF_, ffm_b2, src, E_,
         X1, E_, X1h, E_, TB_, E_, DFF_, 0);

    // --- self-attention ---------------------------------------------------
    gemm(X1h, E_, inh, E_, in_proj_b, nullptr, 0,
         QKV, E3_, nullptr, 0, TB_, E3_, E_, 0);
    {
        const int n = B_ * H_ * T_ * D_;
        repack_qkv_kernel<<<dim3((n + 255) / 256), 256, 0, stream>>>(
            QKV, pos_u, pos_v, Quh, Qvh, Khh, Vth);
    }
    // positional projection P = pos_emb @ wpos^T  (2047 x 512)
    gemm(posh, E_, wposh, E_, nullptr, nullptr, 0,
         nullptr, 0, Ph, E_, PR_, E_, E_, 0);

    for (int b = 0; b < B_; ++b) {
        for (int h = 0; h < H_; ++h) {
            const size_t bh = (size_t)b * H_ + h;
            const bf16_t* Qu = Quh + bh * T_ * D_;
            const bf16_t* Qv = Qvh + bh * T_ * D_;
            const bf16_t* Kk = Khh + bh * T_ * D_;
            const bf16_t* Vt = Vth + bh * (size_t)D_ * T_;
            // ac = (q+u) K^T        (T x T, K-dim 64)
            gemm(Qu, D_, Kk, D_, nullptr, nullptr, 0,
                 acB, T_, nullptr, 0, T_, T_, D_, 0);
            // bd = (q+v) P_h^T      (T x 2047, K-dim 64); P_h rows strided in Ph
            gemm(Qv, D_, Ph + h * D_, E_, nullptr, nullptr, 0,
                 bdB, PR_, nullptr, 0, T_, PR_, D_, 0);
            // rel-shift + softmax -> bf16 probs
            relshift_softmax_kernel<<<dim3(T_), 256, 0, stream>>>(acB, bdB, prB);
            // context = probs @ V   (T x 64, K-dim 1024), scatter into (t*B+b, E)
            gemm(prB, T_, Vt, T_, nullptr, nullptr, 0,
                 nullptr, 0, Ctxh + (size_t)b * E_ + h * D_, B_ * E_,
                 T_, D_, T_, 0);
        }
    }
    // output projection + residual
    gemm(Ctxh, E_, outh, E_, out_proj_b, X1, E_,
         X2, E_, X2h, E_, TB_, E_, E_, 0);

    // --- convolution module ----------------------------------------------
    gemm(X2h, E_, pw1h, E_, pw1_b, nullptr, 0,
         Gbuf, 2 * E_, nullptr, 0, TB_, 2 * E_, E_, 0);
    {
        const int n = TB_ * E_;
        glu_kernel<<<dim3((n + 255) / 256), 256, 0, stream>>>(Gbuf, gluB);
    }
    dwconv_kernel<<<dim3(B_ * E_), 256, 0, stream>>>(gluB, dw_w, dw_b, Y2h);
    gemm(Y2h, E_, pw2h, E_, pw2_b, X2, E_,
         X3, E_, X3h, E_, TB_, E_, E_, 0);

    // --- second FFN -------------------------------------------------------
    gemm(X3h, E_, ff1h, E_, ff_b1, nullptr, 0,
         nullptr, 0, Hh, DFF_, TB_, DFF_, E_, /*dswish*/1);
    gemm(Hh, DFF_, ff2h, DFF_, ff_b2, X3, E_,
         X4, E_, nullptr, 0, TB_, E_, DFF_, 0);

    // --- BasicNorm -> output ---------------------------------------------
    basicnorm_kernel<<<dim3(TB_), 256, 0, stream>>>(X4, norm_eps, out);
}
